// CiaoSR_78134045049274
// MI455X (gfx1250) — matrix operand order repaired and checked by reading, not compiled, not run on hardware
//
#include <hip/hip_runtime.h>
#include <hip/hip_bf16.h>

typedef __bf16 bf16_t;
typedef __attribute__((ext_vector_type(16))) __bf16 v16bf;
typedef __attribute__((ext_vector_type(8)))  float   v8f;
typedef __attribute__((ext_vector_type(8)))  __bf16  bf16x8;
typedef __attribute__((ext_vector_type(4)))  int     v4i_t;

#define GLOBAL_AS __attribute__((address_space(1)))
#define LOCAL_AS  __attribute__((address_space(3)))

namespace {
constexpr int kHF = 48, kWF = 48, kC = 64;
constexpr int kQ  = 96 * 96;        // 9216 queries
constexpr int kCU = kC * 9;         // 576
constexpr int kKinP = 608;          // 580 padded to multiple of 32
constexpr int kHid = 256;
constexpr float kEps = 1e-6f;
}

__device__ __forceinline__ int iclamp(int v, int lo, int hi) {
  return v < lo ? lo : (v > hi ? hi : v);
}

// ---------------------------------------------------------------------------
// 1. 3x3 SAME conv encoder: x[3,48,48] -> feat[64,48,48]  (f32)
// ---------------------------------------------------------------------------
__global__ void conv_enc_kernel(const float* __restrict__ x,
                                const float* __restrict__ W,   // [64,3,3,3] OIHW
                                const float* __restrict__ b,
                                float* __restrict__ feat) {
  int idx = blockIdx.x * blockDim.x + threadIdx.x;
  if (idx >= kC * kHF * kWF) return;
  int c = idx / (kHF * kWF);
  int rem = idx - c * kHF * kWF;
  int y = rem / kWF, xx = rem % kWF;
  float s = b[c];
#pragma unroll
  for (int ic = 0; ic < 3; ++ic)
#pragma unroll
    for (int ki = 0; ki < 3; ++ki)
#pragma unroll
      for (int kj = 0; kj < 3; ++kj) {
        int yy = y + ki - 1, xc = xx + kj - 1;
        if (yy >= 0 && yy < kHF && xc >= 0 && xc < kWF)
          s += x[(ic * kHF + yy) * kWF + xc] * W[((c * 3 + ic) * 3 + ki) * 3 + kj];
      }
  feat[idx] = s;
}

// ---------------------------------------------------------------------------
// 2. Weight convert+transpose: W[K,N] f32 -> Wt[N,Kp] bf16 (zero-pad K->Kp)
// ---------------------------------------------------------------------------
__global__ void wtrans_kernel(const float* __restrict__ W, bf16_t* __restrict__ Wt,
                              int K, int N, int Kp) {
  int idx = blockIdx.x * blockDim.x + threadIdx.x;
  if (idx >= N * Kp) return;
  int n = idx / Kp, k = idx % Kp;
  float v = (k < K) ? W[k * N + n] : 0.f;
  Wt[idx] = (bf16_t)v;
}

// ---------------------------------------------------------------------------
// 3. Gather kernel: builds query [Q,576] (extra=0) or Kin_j [Q,608] (extra=1)
// ---------------------------------------------------------------------------
__global__ void build_kin_kernel(const float* __restrict__ feat,
                                 const float* __restrict__ coord,
                                 const float* __restrict__ cell,
                                 bf16_t* __restrict__ out,
                                 int kp, int extra, float vx, float vy) {
  int idx = blockIdx.x * blockDim.x + threadIdx.x;
  if (idx >= kQ * kp) return;
  int q = idx / kp, col = idx % kp;
  float cy = coord[2 * q], cx = coord[2 * q + 1];
  float cy_ = cy, cx_ = cx;
  if (extra) {
    float c0 = cell[0], c1 = cell[1];
    float rx = (1.f - c0) / (kHF - 1.f);   // 1/tx
    float ry = (1.f - c1) / (kWF - 1.f);   // 1/ty
    cy_ = cy + vx * rx + kEps;
    cx_ = cx + vy * ry + kEps;
    cy_ = fminf(fmaxf(cy_, -1.f + 1e-6f), 1.f - 1e-6f);
    cx_ = fminf(fmaxf(cx_, -1.f + 1e-6f), 1.f - 1e-6f);
  }
  int iy = iclamp((int)rintf(((cy_ + 1.f) * kHF - 1.f) * 0.5f), 0, kHF - 1);
  int ix = iclamp((int)rintf(((cx_ + 1.f) * kWF - 1.f) * 0.5f), 0, kWF - 1);
  float v;
  if (col < kCU) {
    int c = col / 9, t = col % 9;
    int yy = iy + (t / 3) - 1, xx = ix + (t % 3) - 1;
    v = (yy >= 0 && yy < kHF && xx >= 0 && xx < kWF) ? feat[(c * kHF + yy) * kWF + xx]
                                                     : 0.f;
  } else if (col == kCU)     v = (cy - (-1.f + (2.f * iy + 1.f) / kHF)) * kHF;
  else if   (col == kCU + 1) v = (cx - (-1.f + (2.f * ix + 1.f) / kWF)) * kWF;
  else if   (col == kCU + 2) v = cell[2 * q] * kHF;
  else if   (col == kCU + 3) v = cell[2 * q + 1] * kWF;
  else                       v = 0.f;
  out[idx] = (bf16_t)v;
}

// ---------------------------------------------------------------------------
// 4. bf16 WMMA GEMM with LDS-staged weights:
//    C[M,N] = act(A[M,Kp] @ Bt[N,Kp]^T + bias)
//    Block = 8 waves; one 64-column weight chunk async-copied to LDS
//    (GLOBAL_LOAD_ASYNC_TO_LDS_B128, ASYNCcnt); each wave owns a 16x64
//    output strip, A-frag from global, B-frags via ds_load_b128.
// ---------------------------------------------------------------------------
__global__ __launch_bounds__(256)
void gemm_bf16_wmma(const bf16_t* __restrict__ A, const bf16_t* __restrict__ Bt,
                    const float* __restrict__ bias, bf16_t* __restrict__ C,
                    int M, int N, int Kp, int relu) {
  extern __shared__ bf16_t Bs[];   // [64][Kp] bf16, up to 77,824 B
  const int nChunks = N >> 6;
  const int bn = blockIdx.x % nChunks;
  const int bm = blockIdx.x / nChunks;
  const int waveid = (int)(threadIdx.x >> 5);
  const int lane = (int)(threadIdx.x & 31);
  const int n0 = bn << 6;
  const int m0 = (bm * 8 + waveid) << 4;   // 16 rows per wave, 128 per block

  // ---- stage B[n0..n0+63][0..Kp) into LDS (cooperative, 16B granules) ----
  {
    const bf16_t* src = Bt + (size_t)n0 * Kp;   // 64 rows of Bt are contiguous
    const int nvec = (64 * Kp) >> 3;            // # of 16-byte vectors
#if __has_builtin(__builtin_amdgcn_global_load_async_to_lds_b128) && \
    __has_builtin(__builtin_amdgcn_s_wait_asynccnt)
    for (int i = threadIdx.x; i < nvec; i += 256) {
      void* gp = const_cast<void*>((const void*)(src + (size_t)i * 8));
      void* lp = (void*)(Bs + (size_t)i * 8);
      __builtin_amdgcn_global_load_async_to_lds_b128(
          (GLOBAL_AS v4i_t*)gp, (LOCAL_AS v4i_t*)lp, 0, 0);
    }
    __builtin_amdgcn_s_wait_asynccnt(0);
#else
    for (int i = threadIdx.x; i < nvec; i += 256)
      ((bf16x8*)Bs)[i] = ((const bf16x8*)src)[i];
#endif
  }
  __syncthreads();
  if (m0 >= M) return;                      // uniform per wave, after barrier

  const int half = lane >> 4;
  const int l16 = lane & 15;
  const bf16_t* arow = A + (size_t)(m0 + l16) * Kp + (half << 3);   // + half*8
  const bf16_t* brow = Bs + (size_t)l16 * Kp + (half << 3);         // + half*8
  const size_t bstride = (size_t)Kp << 4;   // 16 LDS columns ahead

  v8f acc[4];
#pragma unroll
  for (int t = 0; t < 4; ++t) acc[t] = v8f{};

  for (int k0 = 0; k0 < Kp; k0 += 32) {
    v16bf a;
    {
      bf16x8 lo = *(const bf16x8*)(arow + k0);        // K = half*8 + 0..7
      bf16x8 hi = *(const bf16x8*)(arow + k0 + 16);   // K = 16 + half*8 + 0..7
#pragma unroll
      for (int e = 0; e < 8; ++e) { a[e] = lo[e]; a[e + 8] = hi[e]; }
    }
    const bf16_t* bp = brow + k0;
#pragma unroll
    for (int t = 0; t < 4; ++t) {
      v16bf bfr;
      bf16x8 lo = *(const bf16x8*)(bp);               // K = half*8 + 0..7
      bf16x8 hi = *(const bf16x8*)(bp + 16);          // K = 16 + half*8 + 0..7
#pragma unroll
      for (int e = 0; e < 8; ++e) { bfr[e] = lo[e]; bfr[e + 8] = hi[e]; }
      acc[t] = __builtin_amdgcn_wmma_f32_16x16x32_bf16(
          false, a, false, bfr, (short)0, acc[t], false, false);
      bp += bstride;
    }
  }

#pragma unroll
  for (int t = 0; t < 4; ++t) {
    int col = n0 + (t << 4) + l16;
    float bs = bias[col];
#pragma unroll
    for (int r = 0; r < 8; ++r) {
      float v = acc[t][r] + bs;
      if (relu) v = v > 0.f ? v : 0.f;
      int row = m0 + (half << 3) + r;
      C[(size_t)row * N + col] = (bf16_t)v;
    }
  }
}

// ---------------------------------------------------------------------------
// 5. Per-offset fused elementwise: attn_j[q] = sum_c q*k*wk ; Pv_j = k*wv
// ---------------------------------------------------------------------------
__global__ __launch_bounds__(256)
void attn_pv_kernel(const bf16_t* __restrict__ query, const bf16_t* __restrict__ Kin,
                    const bf16_t* __restrict__ wk, const bf16_t* __restrict__ wv,
                    bf16_t* __restrict__ Pv, float* __restrict__ attn, int j) {
  int wid = (int)((blockIdx.x * blockDim.x + threadIdx.x) >> 5);
  int lane = (int)(threadIdx.x & 31);
  if (wid >= kQ) return;
  const bf16_t* qr  = query + (size_t)wid * kCU;
  const bf16_t* kr  = Kin + (size_t)wid * kKinP;
  const bf16_t* wkr = wk + (size_t)wid * kCU;
  const bf16_t* wvr = wv + (size_t)wid * kCU;
  bf16_t* pvr = Pv + (size_t)wid * kCU;
  float s = 0.f;
  for (int c = lane; c < kCU; c += 32) {
    float kf = (float)kr[c];
    s += (float)qr[c] * kf * (float)wkr[c];
    pvr[c] = (bf16_t)(kf * (float)wvr[c]);
  }
#pragma unroll
  for (int off = 16; off > 0; off >>= 1) s += __shfl_xor(s, off, 32);
  if (lane == 0) attn[wid * 4 + j] = s;
}

// 6. softmax over the 4 offsets
__global__ void softmax4_kernel(const float* __restrict__ attn, float* __restrict__ probs) {
  int q = blockIdx.x * blockDim.x + threadIdx.x;
  if (q >= kQ) return;
  float a0 = attn[4 * q], a1 = attn[4 * q + 1], a2 = attn[4 * q + 2], a3 = attn[4 * q + 3];
  float m = fmaxf(fmaxf(a0, a1), fmaxf(a2, a3));
  float e0 = expf(a0 - m), e1 = expf(a1 - m), e2 = expf(a2 - m), e3 = expf(a3 - m);
  float inv = 1.f / (e0 + e1 + e2 + e3);
  probs[4 * q] = e0 * inv; probs[4 * q + 1] = e1 * inv;
  probs[4 * q + 2] = e2 * inv; probs[4 * q + 3] = e3 * inv;
}

// 7. out[q,c] = sum_j p_j * Pv_j[q,c]   -> bf16 for final GEMM
__global__ void combine_kernel(const float* __restrict__ probs, const bf16_t* __restrict__ Pv,
                               bf16_t* __restrict__ outm) {
  int idx = blockIdx.x * blockDim.x + threadIdx.x;
  if (idx >= kQ * kCU) return;
  int q = idx / kCU;
  const size_t stride = (size_t)kQ * kCU;
  float v = probs[4 * q]     * (float)Pv[idx]
          + probs[4 * q + 1] * (float)Pv[idx + stride]
          + probs[4 * q + 2] * (float)Pv[idx + 2 * stride]
          + probs[4 * q + 3] * (float)Pv[idx + 3 * stride];
  outm[idx] = (bf16_t)v;
}

// 8. final 256->3 layer + bilinear-border residual of x
__global__ void final_head_kernel(const bf16_t* __restrict__ H3,
                                  const float* __restrict__ Wq2, const float* __restrict__ bq2,
                                  const float* __restrict__ x, const float* __restrict__ coord,
                                  float* __restrict__ out) {
  int q = blockIdx.x * blockDim.x + threadIdx.x;
  if (q >= kQ) return;
  float a0 = bq2[0], a1 = bq2[1], a2 = bq2[2];
  const bf16_t* h = H3 + (size_t)q * kHid;
  for (int k = 0; k < kHid; ++k) {
    float hv = (float)h[k];
    a0 += hv * Wq2[3 * k];
    a1 += hv * Wq2[3 * k + 1];
    a2 += hv * Wq2[3 * k + 2];
  }
  float cy = coord[2 * q], cx = coord[2 * q + 1];
  float fy = ((cy + 1.f) * kHF - 1.f) * 0.5f;
  float fx = ((cx + 1.f) * kWF - 1.f) * 0.5f;
  float y0f = floorf(fy), x0f = floorf(fx);
  float wy = fy - y0f, wx = fx - x0f;
  int y0 = iclamp((int)y0f, 0, kHF - 1), y1 = iclamp((int)y0f + 1, 0, kHF - 1);
  int x0 = iclamp((int)x0f, 0, kWF - 1), x1 = iclamp((int)x0f + 1, 0, kWF - 1);
  float res[3] = {a0, a1, a2};
#pragma unroll
  for (int ch = 0; ch < 3; ++ch) {
    const float* img = x + ch * kHF * kWF;
    float b = img[y0 * kWF + x0] * (1.f - wy) * (1.f - wx)
            + img[y0 * kWF + x1] * (1.f - wy) * wx
            + img[y1 * kWF + x0] * wy * (1.f - wx)
            + img[y1 * kWF + x1] * wy * wx;
    out[3 * q + ch] = res[ch] + b;
  }
}

// ---------------------------------------------------------------------------
extern "C" void kernel_launch(void* const* d_in, const int* in_sizes, int n_in,
                              void* d_out, int out_size, void* d_ws, size_t ws_size,
                              hipStream_t stream) {
  (void)in_sizes; (void)n_in; (void)out_size; (void)ws_size;
  const float* x     = (const float*)d_in[0];
  const float* coord = (const float*)d_in[1];
  const float* cell  = (const float*)d_in[2];
  const float* W_enc = (const float*)d_in[3];
  const float* b_enc = (const float*)d_in[4];
  const float* Wk1 = (const float*)d_in[5];
  const float* bk1 = (const float*)d_in[6];
  const float* Wk2 = (const float*)d_in[7];
  const float* bk2 = (const float*)d_in[8];
  const float* Wv1 = (const float*)d_in[9];
  const float* bv1 = (const float*)d_in[10];
  const float* Wv2 = (const float*)d_in[11];
  const float* bv2 = (const float*)d_in[12];
  const float* Wq1 = (const float*)d_in[13];
  const float* bq1 = (const float*)d_in[14];
  const float* Wq2 = (const float*)d_in[15];
  const float* bq2 = (const float*)d_in[16];
  float* out = (float*)d_out;
  char* ws = (char*)d_ws;

  // workspace layout (all 256B-aligned)
  float*  feat  = (float*)(ws + 0);                 //   589,824 B
  bf16_t* Wk1t  = (bf16_t*)(ws + 589824);           //   311,296 B  [256,608]
  bf16_t* Wk2t  = (bf16_t*)(ws + 901120);           //   294,912 B  [576,256]
  bf16_t* Wv1t  = (bf16_t*)(ws + 1196032);          //   311,296 B
  bf16_t* Wv2t  = (bf16_t*)(ws + 1507328);          //   294,912 B
  bf16_t* Wq1t  = (bf16_t*)(ws + 1802240);          //   294,912 B  [256,576]
  bf16_t* qry   = (bf16_t*)(ws + 2097152);          // 10,616,832 B [Q,576]
  bf16_t* Kin   = (bf16_t*)(ws + 12713984);         // 11,206,656 B [Q,608]
  bf16_t* Hbuf  = (bf16_t*)(ws + 23920640);         //  4,718,592 B [Q,256]
  bf16_t* wkS   = (bf16_t*)(ws + 28639232);         // 10,616,832 B [Q,576]
  bf16_t* wvS   = (bf16_t*)(ws + 39256064);         // 10,616,832 B
  bf16_t* Pv    = (bf16_t*)(ws + 49872896);         // 42,467,328 B [4][Q,576]
  float*  attn  = (float*)(ws + 92340224);          //    147,456 B [Q,4]
  float*  probs = (float*)(ws + 92487680);          //    147,456 B
  bf16_t* outm  = (bf16_t*)(ws + 92635136);         // 10,616,832 B [Q,576]
  bf16_t* H3    = (bf16_t*)(ws + 103251968);        //  4,718,592 B [Q,256]

  auto blocks = [](long n) { return (unsigned)((n + 255) / 256); };
  auto gemm = [&](const bf16_t* A, const bf16_t* Bt, const float* bias, bf16_t* C,
                  int M, int N, int Kp, int relu) {
    int nblk = (M / 128) * (N >> 6);           // 8 waves x 16 rows = 128 rows/block
    size_t lds = (size_t)64 * Kp * sizeof(bf16_t);
    gemm_bf16_wmma<<<nblk, 256, lds, stream>>>(A, Bt, bias, C, M, N, Kp, relu);
  };

  // encoder + weight prep
  conv_enc_kernel<<<blocks(kC * kHF * kWF), 256, 0, stream>>>(x, W_enc, b_enc, feat);
  wtrans_kernel<<<blocks(256 * kKinP), 256, 0, stream>>>(Wk1, Wk1t, 580, 256, kKinP);
  wtrans_kernel<<<blocks(576 * 256),  256, 0, stream>>>(Wk2, Wk2t, 256, 576, 256);
  wtrans_kernel<<<blocks(256 * kKinP), 256, 0, stream>>>(Wv1, Wv1t, 580, 256, kKinP);
  wtrans_kernel<<<blocks(576 * 256),  256, 0, stream>>>(Wv2, Wv2t, 256, 576, 256);
  wtrans_kernel<<<blocks(256 * 576),  256, 0, stream>>>(Wq1, Wq1t, 576, 256, 576);

  // query gather (no shift)
  build_kin_kernel<<<blocks((long)kQ * kCU), 256, 0, stream>>>(
      feat, coord, cell, qry, kCU, 0, 0.f, 0.f);

  // 4 diagonal offsets, order matches reference (vx outer, vy inner)
  const float vxs[4] = {-1.f, -1.f, 1.f, 1.f};
  const float vys[4] = {-1.f, 1.f, -1.f, 1.f};
  for (int j = 0; j < 4; ++j) {
    build_kin_kernel<<<blocks((long)kQ * kKinP), 256, 0, stream>>>(
        feat, coord, cell, Kin, kKinP, 1, vxs[j], vys[j]);
    gemm(Kin,  Wk1t, bk1, Hbuf, kQ, 256, kKinP, 1);   // key MLP layer 1 (ReLU)
    gemm(Hbuf, Wk2t, bk2, wkS,  kQ, 576, 256,   0);   // key MLP layer 2
    gemm(Kin,  Wv1t, bv1, Hbuf, kQ, 256, kKinP, 1);   // value MLP layer 1
    gemm(Hbuf, Wv2t, bv2, wvS,  kQ, 576, 256,   0);   // value MLP layer 2
    attn_pv_kernel<<<(kQ * 32 + 255) / 256, 256, 0, stream>>>(
        qry, Kin, wkS, wvS, Pv + (size_t)j * kQ * kCU, attn, j);
  }

  softmax4_kernel<<<blocks(kQ), 256, 0, stream>>>(attn, probs);
  combine_kernel<<<blocks((long)kQ * kCU), 256, 0, stream>>>(probs, Pv, outm);

  gemm(outm, Wq1t, bq1, H3, kQ, 256, 576, 1);          // head layer 1 (ReLU)
  final_head_kernel<<<blocks(kQ), 256, 0, stream>>>(H3, Wq2, bq2, x, coord, out);
}
